// MAF_Extractor_78288663872200
// MI455X (gfx1250) — matrix operand order repaired and checked by reading, not compiled, not run on hardware
//
#include <hip/hip_runtime.h>

// ---------------------------------------------------------------------------
// MAF extractor: fused grid_sample + 3-layer point MLP (WMMA bf16) + final FC
// Target: gfx1250 (CDNA5, wave32, v_wmma_f32_16x16x32_bf16)
// Software-pipelined: next iteration's A/B fragment loads are issued before
// the current WMMA so s_wait_loadcnt no longer drains to zero per matrix op.
// ---------------------------------------------------------------------------

typedef __attribute__((ext_vector_type(16))) __bf16 v16bf;
typedef __attribute__((ext_vector_type(8)))  __bf16 v8bf;
typedef __attribute__((ext_vector_type(8)))  float  v8f;

#define BATCH 64
#define NPTS  500
#define CCH   256
#define HH    64
#define WW    64

union ABf { v16bf v; v8bf h[2]; };

// Build a 16x32 bf16 A-fragment for one lane.
// A layout (matches contiguous B fill): lane half=0 carries K {k0..k0+15},
// half=1 carries K {k0+16..k0+31}; row M = lane&15.
// rowp points at (&buf[M*stride + k0]).
static __device__ __forceinline__ v16bf load_a_frag(const __bf16* rowp, int hf) {
    ABf u;
    u.h[0] = *(const v8bf*)(rowp + 16 * hf);
    u.h[1] = *(const v8bf*)(rowp + 16 * hf + 8);
    return u.v;
}

static __device__ __forceinline__ v8f wmma_bf16(v16bf a, v16bf b, v8f c) {
    return __builtin_amdgcn_wmma_f32_16x16x32_bf16(false, a, false, b,
                                                   (short)0, c, false, false);
}

// ---------------------------------------------------------------------------
// Kernel 1: convert weights fp32 -> bf16 into workspace (w2 zero-padded 5->16)
// ---------------------------------------------------------------------------
__global__ void prep_kernel(const float* __restrict__ w0,
                            const float* __restrict__ w1,
                            const float* __restrict__ w2,
                            __bf16* __restrict__ w0b,
                            __bf16* __restrict__ w1b,
                            __bf16* __restrict__ w2b) {
    int i = blockIdx.x * 256 + threadIdx.x;
    if (i < 128 * 256) w0b[i] = (__bf16)w0[i];
    if (i < 64 * 384)  w1b[i] = (__bf16)w1[i];
    if (i < 16 * 320) {
        int r = i / 320;
        w2b[i] = (r < 5) ? (__bf16)w2[i] : (__bf16)0.0f;
    }
}

// ---------------------------------------------------------------------------
// Kernel 2: fused gather + MLP. 4 waves per block, 1 wave == 16 points.
// ---------------------------------------------------------------------------
__global__ __launch_bounds__(128) void maf_main(
    const float* __restrict__ p,       // [B,N,2]
    const float* __restrict__ s_feat,  // [B,C,H,W]
    const float* __restrict__ b0g,     // [128]
    const float* __restrict__ b1g,     // [64]
    const float* __restrict__ b2g,     // [5]
    const __bf16* __restrict__ w0b,    // [128,256]
    const __bf16* __restrict__ w1b,    // [64,384]
    const __bf16* __restrict__ w2b,    // [16,320] (rows>=5 zero)
    float* __restrict__ yout)          // [B, 5*N]  (o*N + n)
{
    // per-wave private LDS: concat1 = [y0(128) | feat(256)] = 16x384,
    // plus y1 = 16x64. 4 waves -> 56 KB static.
    __shared__ __align__(32) __bf16 s_cat1[4][16 * 384];
    __shared__ __align__(32) __bf16 s_y1[4][16 * 64];

    const int b    = blockIdx.x;
    const int wave = threadIdx.x >> 5;
    const int lane = threadIdx.x & 31;
    const int col  = lane & 15;   // WMMA N-column / A row M
    const int hf   = lane >> 4;   // lane half
    const int M    = col;         // A-matrix row for this lane
    const int tile_base = blockIdx.y * 64 + wave * 16;

    __bf16* cat1 = s_cat1[wave];
    __bf16* y1b  = s_y1[wave];

    // ---------------- bilinear gather into cat1[:,128:384] -----------------
    const float* img = s_feat + (size_t)b * (CCH * HH * WW);
    for (int pi = 0; pi < 16; ++pi) {
        int n = tile_base + pi;
        float vx = 0.0f, vy = 0.0f, valid = 0.0f;
        if (n < NPTS) {
            vx = p[((size_t)b * NPTS + n) * 2 + 0];
            vy = p[((size_t)b * NPTS + n) * 2 + 1];
            valid = 1.0f;
        }
        // align_corners=True mapping
        float xf = (vx + 1.0f) * 0.5f * (float)(WW - 1);
        float yf = (vy + 1.0f) * 0.5f * (float)(HH - 1);
        float x0f = floorf(xf), y0f = floorf(yf);
        float wx1 = xf - x0f, wx0 = 1.0f - wx1;
        float wy1 = yf - y0f, wy0 = 1.0f - wy1;
        int ix0 = (int)x0f, iy0 = (int)y0f;
        int ix1 = ix0 + 1,  iy1 = iy0 + 1;
        float mx0 = (ix0 >= 0 && ix0 < WW) ? 1.0f : 0.0f;
        float mx1 = (ix1 >= 0 && ix1 < WW) ? 1.0f : 0.0f;
        float my0 = (iy0 >= 0 && iy0 < HH) ? 1.0f : 0.0f;
        float my1 = (iy1 >= 0 && iy1 < HH) ? 1.0f : 0.0f;
        int cx0 = min(max(ix0, 0), WW - 1), cx1 = min(max(ix1, 0), WW - 1);
        int cy0 = min(max(iy0, 0), HH - 1), cy1 = min(max(iy1, 0), HH - 1);
        int i00 = cy0 * WW + cx0, i10 = cy0 * WW + cx1;
        int i01 = cy1 * WW + cx0, i11 = cy1 * WW + cx1;
        float w00 = valid * wx0 * wy0 * mx0 * my0;
        float w10 = valid * wx1 * wy0 * mx1 * my0;
        float w01 = valid * wx0 * wy1 * mx0 * my1;
        float w11 = valid * wx1 * wy1 * mx1 * my1;
        #pragma unroll
        for (int j = 0; j < 8; ++j) {
            int c = lane + 32 * j;
            const float* pl = img + (size_t)c * (HH * WW);
            float v = w00 * pl[i00] + w10 * pl[i10] +
                      w01 * pl[i01] + w11 * pl[i11];
            cat1[pi * 384 + 128 + c] = (__bf16)v;
        }
    }
    // waves own disjoint LDS regions; no barrier needed (DS ops in-order per wave)

    // ---------------- MLP0: [16x256] x [256x128] -> y0 ----------------------
    // 64 flattened (ks,t) steps, 1-deep register rotation on A and B frags.
    v8f acc0[8];
    #pragma unroll
    for (int t = 0; t < 8; ++t)
        #pragma unroll
        for (int v = 0; v < 8; ++v) acc0[t][v] = 0.0f;

    {
        const __bf16* arow = cat1 + M * 384 + 128;
        v16bf a_cur = load_a_frag(arow, hf);
        v16bf b_cur = *(const v16bf*)(w0b + (size_t)col * 256 + 16 * hf);
        #pragma unroll
        for (int i = 0; i < 64; ++i) {
            const int t  = i & 7;
            const int ni = i + 1;
            v16bf a_nxt = a_cur;
            v16bf b_nxt = b_cur;
            if (ni < 64) {
                const int ksn = ni >> 3, tn = ni & 7;
                if (tn == 0) a_nxt = load_a_frag(arow + ksn * 32, hf);
                b_nxt = *(const v16bf*)(w0b + (size_t)(tn * 16 + col) * 256 +
                                        ksn * 32 + 16 * hf);
            }
            acc0[t] = wmma_bf16(a_cur, b_cur, acc0[t]);
            a_cur = a_nxt;
            b_cur = b_nxt;
        }
    }
    #pragma unroll
    for (int t = 0; t < 8; ++t) {
        float bias = b0g[t * 16 + col];
        #pragma unroll
        for (int v = 0; v < 8; ++v) {
            float x = acc0[t][v] + bias;
            x = (x > 0.0f) ? x : 0.01f * x;          // leaky_relu
            cat1[(v + 8 * hf) * 384 + t * 16 + col] = (__bf16)x;
        }
    }

    // ---------------- MLP1: [16x384] x [384x64] -> y1 -----------------------
    v8f acc1[4];
    #pragma unroll
    for (int t = 0; t < 4; ++t)
        #pragma unroll
        for (int v = 0; v < 8; ++v) acc1[t][v] = 0.0f;

    {
        const __bf16* arow = cat1 + M * 384;
        v16bf a_cur = load_a_frag(arow, hf);
        v16bf b_cur = *(const v16bf*)(w1b + (size_t)col * 384 + 16 * hf);
        #pragma unroll
        for (int i = 0; i < 48; ++i) {               // K = 384, 4 N-tiles
            const int t  = i & 3;
            const int ni = i + 1;
            v16bf a_nxt = a_cur;
            v16bf b_nxt = b_cur;
            if (ni < 48) {
                const int ksn = ni >> 2, tn = ni & 3;
                if (tn == 0) a_nxt = load_a_frag(arow + ksn * 32, hf);
                b_nxt = *(const v16bf*)(w1b + (size_t)(tn * 16 + col) * 384 +
                                        ksn * 32 + 16 * hf);
            }
            acc1[t] = wmma_bf16(a_cur, b_cur, acc1[t]);
            a_cur = a_nxt;
            b_cur = b_nxt;
        }
    }
    #pragma unroll
    for (int t = 0; t < 4; ++t) {
        float bias = b1g[t * 16 + col];
        #pragma unroll
        for (int v = 0; v < 8; ++v) {
            float x = acc1[t][v] + bias;
            x = (x > 0.0f) ? x : 0.01f * x;
            y1b[(v + 8 * hf) * 64 + t * 16 + col] = (__bf16)x;
        }
    }

    // ---------------- MLP2: [16x320] x [320x16(5)] -> y2 --------------------
    v8f acc2;
    #pragma unroll
    for (int v = 0; v < 8; ++v) acc2[v] = 0.0f;

    {
        // K layout: steps 0..1 from y1 (cols 0..63), steps 2..9 from feat
        // region of cat1 (concat boundary aligns with the 32-wide K step).
        v16bf a_cur = load_a_frag(y1b + M * 64, hf);
        v16bf b_cur = *(const v16bf*)(w2b + (size_t)col * 320 + 16 * hf);
        #pragma unroll
        for (int ks = 0; ks < 10; ++ks) {
            const int ni = ks + 1;
            v16bf a_nxt = a_cur;
            v16bf b_nxt = b_cur;
            if (ni < 10) {
                const __bf16* ap = (ni < 2) ? (y1b + M * 64 + ni * 32)
                                            : (cat1 + M * 384 + ni * 32 + 64);
                a_nxt = load_a_frag(ap, hf);
                b_nxt = *(const v16bf*)(w2b + (size_t)col * 320 +
                                        ni * 32 + 16 * hf);
            }
            acc2 = wmma_bf16(a_cur, b_cur, acc2);
            a_cur = a_nxt;
            b_cur = b_nxt;
        }
    }
    float bias2 = (col < 5) ? b2g[col] : 0.0f;
    #pragma unroll
    for (int v = 0; v < 8; ++v) {
        int n = tile_base + v + 8 * hf;
        float x = acc2[v] + bias2;
        x = (x > 0.0f) ? x : 0.0f;                   // relu
        if (col < 5 && n < NPTS)
            yout[(size_t)b * (5 * NPTS) + col * NPTS + n] = x;
    }
}

// ---------------------------------------------------------------------------
// Kernel 3: final FC  out[b,j] = sum_k y[b,k]*fc_w[j,k] + fc_b[j]
// ---------------------------------------------------------------------------
__global__ __launch_bounds__(32) void fc_kernel(const float* __restrict__ y,
                                                const float* __restrict__ fcw,
                                                const float* __restrict__ fcb,
                                                float* __restrict__ out) {
    int j = blockIdx.x;            // 0..235
    int b = blockIdx.y;            // 0..63
    int lane = threadIdx.x;
    const float* yr = y + (size_t)b * (5 * NPTS);
    const float* wr = fcw + (size_t)j * (5 * NPTS);
    float s = 0.0f;
    for (int k = lane; k < 5 * NPTS; k += 32) s += yr[k] * wr[k];
    #pragma unroll
    for (int off = 16; off > 0; off >>= 1) s += __shfl_down(s, off, 32);
    if (lane == 0) out[(size_t)b * 236 + j] = s + fcb[j];
}

// ---------------------------------------------------------------------------
extern "C" void kernel_launch(void* const* d_in, const int* in_sizes, int n_in,
                              void* d_out, int out_size, void* d_ws, size_t ws_size,
                              hipStream_t stream) {
    const float* p   = (const float*)d_in[0];
    const float* sf  = (const float*)d_in[1];
    const float* w0  = (const float*)d_in[2];
    const float* b0  = (const float*)d_in[3];
    const float* w1  = (const float*)d_in[4];
    const float* b1  = (const float*)d_in[5];
    const float* w2  = (const float*)d_in[6];
    const float* b2  = (const float*)d_in[7];
    const float* fcw = (const float*)d_in[8];
    const float* fcb = (const float*)d_in[9];
    float* out = (float*)d_out;

    char* ws = (char*)d_ws;
    __bf16* w0b = (__bf16*)(ws);                    // 128*256*2 = 65536 B
    __bf16* w1b = (__bf16*)(ws + 65536);            //  64*384*2 = 49152 B
    __bf16* w2b = (__bf16*)(ws + 65536 + 49152);    //  16*320*2 = 10240 B
    float*  ybuf = (float*)(ws + 131072);           // 64*2500*4 = 640000 B

    hipLaunchKernelGGL(prep_kernel, dim3(128), dim3(256), 0, stream,
                       w0, w1, w2, w0b, w1b, w2b);
    hipLaunchKernelGGL(maf_main, dim3(BATCH, 8), dim3(128), 0, stream,
                       p, sf, b0, b1, b2, w0b, w1b, w2b, ybuf);
    hipLaunchKernelGGL(fc_kernel, dim3(236, BATCH), dim3(32), 0, stream,
                       ybuf, fcw, fcb, out);
}
